// TransfoXL_82935818485910
// MI455X (gfx1250) — compile-verified, matrix operand order checked by reading, not run on hardware
//
#include <hip/hip_runtime.h>

// ---------------------------------------------------------------------------
// Transformer-XL attention layer for MI455X (gfx1250, wave32, WMMA bf16).
// Flash-attention formulation (no materialized score tensor), bf16 WMMA
// 16x16x32 with fp32 accumulation for all four GEMMs; softmax/LN in fp32.
// ---------------------------------------------------------------------------

typedef __bf16 bf16;
typedef __attribute__((ext_vector_type(16))) __bf16 v16bf;
typedef __attribute__((ext_vector_type(8)))  __bf16 v8bf;
typedef __attribute__((ext_vector_type(8)))  float  v8f;

static constexpr int cQLEN = 1024;
static constexpr int cMLEN = 1024;
static constexpr int cKLEN = 2048;
static constexpr int cBSZ  = 4;
static constexpr int cNH   = 16;
static constexpr int cDH   = 64;
static constexpr int cDM   = 1024;   // d_model
static constexpr int cQKV  = 3072;   // 3 * NH * DH

__device__ __forceinline__ v8f vzero8() {
  v8f z;
#pragma unroll
  for (int e = 0; e < 8; ++e) z[e] = 0.0f;
  return z;
}

__device__ __forceinline__ v8f wmma_bf16(v16bf a, v16bf b, v8f c) {
  return __builtin_amdgcn_wmma_f32_16x16x32_bf16(
      /*neg_a=*/false, a, /*neg_b=*/false, b,
      /*c_mod=*/(short)0, c, /*reuse_a=*/false, /*reuse_b=*/false);
}

// Load a 16x32 WMMA operand (A layout, or B layout when memory is [N][K])
// from bf16 memory with row stride `stride` (elements). Per ISA layout:
// lane l: row = l&15, half h = l>>4; elems 0..7 = K 8h..8h+7,
// elems 8..15 = K 16+8h..16+8h+7  -> two contiguous 16B loads.
__device__ __forceinline__ v16bf wmma_op_load_bf16(const bf16* base, int stride) {
  const int l = threadIdx.x & 31;
  const int r = l & 15, h = l >> 4;
  const bf16* p = base + (size_t)r * stride + 8 * h;
  union { v8bf half[2]; v16bf v; } u;
  u.half[0] = *reinterpret_cast<const v8bf*>(p);
  u.half[1] = *reinterpret_cast<const v8bf*>(p + 16);
  return u.v;
}

// Same, but source is fp32 (converted to bf16 in registers).
__device__ __forceinline__ v16bf wmma_opA_load_f32(const float* base, int stride) {
  const int l = threadIdx.x & 31;
  const int r = l & 15, h = l >> 4;
  const float* p = base + (size_t)r * stride + 8 * h;
  v16bf v;
#pragma unroll
  for (int e = 0; e < 8; ++e) v[e] = (__bf16)p[e];
#pragma unroll
  for (int e = 0; e < 8; ++e) v[8 + e] = (__bf16)p[16 + e];
  return v;
}

// B operand for the relative-position window GEMM: N index = clamped m row of
// r_emb (layout [m][nh][dh], bf16), K = head dim chunk starting at kk.
__device__ __forceinline__ v16bf wmma_opB_load_rel(const bf16* rbase, int m0, int kk) {
  const int l = threadIdx.x & 31;
  const int h = l >> 4;
  int m = m0 + (l & 15);
  m = m < 0 ? 0 : (m > (cKLEN - 1) ? (cKLEN - 1) : m);
  const bf16* p = rbase + (size_t)m * (cNH * cDH) + kk + 8 * h;
  union { v8bf half[2]; v16bf v; } u;
  u.half[0] = *reinterpret_cast<const v8bf*>(p);
  u.half[1] = *reinterpret_cast<const v8bf*>(p + 16);
  return u.v;
}

// ---------------------------------------------------------------------------
// Kernel 1: tiled transpose + fp32->bf16 convert:  src[K][N] -> dst[N][K]
// ---------------------------------------------------------------------------
__global__ __launch_bounds__(256) void transpose_cvt_kernel(
    const float* __restrict__ src, bf16* __restrict__ dst, int K, int N) {
  __shared__ float tile[32][33];
  const int n0 = blockIdx.x * 32, k0 = blockIdx.y * 32;
  const int tx = threadIdx.x & 31, ty = threadIdx.x >> 5;  // 32x8
#pragma unroll
  for (int yy = ty; yy < 32; yy += 8)
    tile[yy][tx] = src[(size_t)(k0 + yy) * N + n0 + tx];
  __syncthreads();
#pragma unroll
  for (int yy = ty; yy < 32; yy += 8)
    dst[(size_t)(n0 + yy) * K + k0 + tx] = (__bf16)tile[tx][yy];
}

// ---------------------------------------------------------------------------
// Kernel 2: elementwise fp32 -> bf16
// ---------------------------------------------------------------------------
__global__ __launch_bounds__(256) void cvt_bf16_kernel(
    const float* __restrict__ src, bf16* __restrict__ dst, int n) {
  int i = blockIdx.x * 256 + threadIdx.x;
  if (i < n) dst[i] = (__bf16)src[i];
}

// ---------------------------------------------------------------------------
// Kernel 3: QKV GEMM.  A = concat(mems, w)  [8192 x 1024] fp32 (row = j*4+b),
// B = WqkvT bf16 [3072][1024].  Wave tile: 16M x 64N, K-loop step 32.
// Epilogue scatters into q / rw_q (natural [b][n][i][d]), k ([b][n][j][d]),
// vT ([b][n][d][j]) as bf16.
// ---------------------------------------------------------------------------
__global__ __launch_bounds__(128) void qkv_gemm_kernel(
    const float* __restrict__ w, const float* __restrict__ mems,
    const bf16* __restrict__ wqkvT, const float* __restrict__ r_w_bias,
    bf16* __restrict__ q_bf, bf16* __restrict__ rwq_bf,
    bf16* __restrict__ k_bf, bf16* __restrict__ vT_bf) {
  const int wid = blockIdx.x * 4 + (threadIdx.x >> 5);
  const int nt = wid % (cQKV / 64);
  const int mt = wid / (cQKV / 64);
  const int r0 = mt * 16, c0 = nt * 64;
  const int lane = threadIdx.x & 31, cl = lane & 15, h = lane >> 4;

  const float* A = (r0 < cMLEN * cBSZ)
                       ? (mems + (size_t)r0 * cDM)
                       : (w + (size_t)(r0 - cMLEN * cBSZ) * cDM);
  v8f acc[4];
#pragma unroll
  for (int u = 0; u < 4; ++u) acc[u] = vzero8();

  for (int kk = 0; kk < cDM; kk += 32) {
    v16bf a = wmma_opA_load_f32(A + kk, cDM);
#pragma unroll
    for (int u = 0; u < 4; ++u) {
      v16bf bb = wmma_op_load_bf16(wqkvT + (size_t)(c0 + 16 * u) * cDM + kk, cDM);
      acc[u] = wmma_bf16(a, bb, acc[u]);
    }
  }

  const int sec = c0 >> 10;            // 0=q, 1=k, 2=v
  const int n = (c0 & 1023) >> 6;      // head index (tile lies in one head)
#pragma unroll
  for (int u = 0; u < 4; ++u) {
#pragma unroll
    for (int g = 0; g < 8; ++g) {
      const int r = r0 + g + 8 * h;    // global A row
      const int j = r >> 2, b = r & 3;
      const int d = 16 * u + cl;
      const float val = acc[u][g];
      const size_t bn = (size_t)(b * cNH + n);
      if (sec == 0) {
        if (j >= cMLEN) {
          const int i = j - cMLEN;
          const size_t off = (bn * cQLEN + i) * cDH + d;
          q_bf[off] = (__bf16)val;
          rwq_bf[off] = (__bf16)(val + r_w_bias[n * cDH + d]);
        }
      } else if (sec == 1) {
        k_bf[(bn * cKLEN + j) * cDH + d] = (__bf16)val;
      } else {
        vT_bf[(bn * cDH + d) * cKLEN + j] = (__bf16)val;
      }
    }
  }
}

// ---------------------------------------------------------------------------
// Kernel 4: flash attention with relative position.  1024 blocks of 128
// threads; 4 independent waves per block; wave owns 16 q-rows of one (b,n).
// Per 32-wide j-step: 4 WMMA (AC) + 6 WMMA (rel window) + 4 WMMA (P@V).
// ---------------------------------------------------------------------------
__global__ __launch_bounds__(128) void attn_kernel(
    const bf16* __restrict__ q_bf, const bf16* __restrict__ rwq_bf,
    const bf16* __restrict__ k_bf, const bf16* __restrict__ vT_bf,
    const bf16* __restrict__ remb_bf, const float* __restrict__ r_bias,
    bf16* __restrict__ avec) {
  __shared__ float s_rbias[cKLEN];
  __shared__ float s_bw[4][16 * 48];
  __shared__ bf16  s_p[4][16 * 32];

  const int wv = threadIdx.x >> 5;
  const int lane = threadIdx.x & 31;
  const int cl = lane & 15, h = lane >> 4;

  const int blk = blockIdx.x;
  const int ib = blk & 15;           // 16 I-blocks of 64 rows
  const int n  = (blk >> 4) & 15;
  const int b  = blk >> 8;
  const int Iw = ib * 64 + wv * 16;  // this wave's q-row base

  for (int m = threadIdx.x; m < cKLEN; m += 128)
    s_rbias[m] = r_bias[m * cNH + n];
  __syncthreads();

  const size_t bn = (size_t)(b * cNH + n);
  const bf16* qb   = q_bf   + (bn * cQLEN + Iw) * cDH;
  const bf16* rwqb = rwq_bf + (bn * cQLEN + Iw) * cDH;
  const bf16* kb   = k_bf   + bn * cKLEN * cDH;
  const bf16* vtb  = vT_bf  + bn * cDH * cKLEN;
  const bf16* rb   = remb_bf + n * cDH;

  const v16bf qA0 = wmma_op_load_bf16(qb, cDH);
  const v16bf qA1 = wmma_op_load_bf16(qb + 32, cDH);
  const v16bf rA0 = wmma_op_load_bf16(rwqb, cDH);
  const v16bf rA1 = wmma_op_load_bf16(rwqb + 32, cDH);

  v8f O[4];
  float mrow[8], lrow[8];
#pragma unroll
  for (int u = 0; u < 4; ++u) O[u] = vzero8();
#pragma unroll
  for (int g = 0; g < 8; ++g) { mrow[g] = -3.4028e38f; lrow[g] = 0.0f; }

  const int jlim = Iw + 16 + cMLEN;               // need j <= i + MLEN
  const int Jmax = (cKLEN < jlim) ? cKLEN : jlim;
  const float scale = 0.125f;                     // 1/sqrt(64)

  float* bw = s_bw[wv];
  bf16* pbuf = s_p[wv];

  for (int J = 0; J < Jmax; J += 32) {
    // -------- AC = rw_q . k^T : two 16x16 subtiles, K=64
    v8f s[2];
#pragma unroll
    for (int t = 0; t < 2; ++t) {
      const bf16* kt = kb + (size_t)(J + 16 * t) * cDH;
      v8f acc = vzero8();
      acc = wmma_bf16(rA0, wmma_op_load_bf16(kt, cDH), acc);
      acc = wmma_bf16(rA1, wmma_op_load_bf16(kt + 32, cDH), acc);
      s[t] = acc;
    }

    // -------- rel window: Bw[i, m0+0..47] = q . r_emb^T, staged to LDS
    const int m0 = J - Iw + 1008;
#pragma unroll
    for (int t = 0; t < 3; ++t) {
      v8f acc = vzero8();
      acc = wmma_bf16(qA0, wmma_opB_load_rel(rb, m0 + 16 * t, 0), acc);
      acc = wmma_bf16(qA1, wmma_opB_load_rel(rb, m0 + 16 * t, 32), acc);
#pragma unroll
      for (int g = 0; g < 8; ++g)
        bw[(g + 8 * h) * 48 + 16 * t + cl] = acc[g];
    }
    asm volatile("s_wait_dscnt 0x0" ::: "memory");

    // -------- score = (AC + shifted Bw + r_bias) * scale, causal mask
#pragma unroll
    for (int t = 0; t < 2; ++t) {
#pragma unroll
      for (int g = 0; g < 8; ++g) {
        const int i_l = g + 8 * h;
        const int j_l = 16 * t + cl;
        const int rel = j_l - i_l + 15;            // in [0, 46]
        const int m = m0 + rel;                    // = j - i + 1023
        const int mc = m < 0 ? 0 : (m > cKLEN - 1 ? cKLEN - 1 : m);
        float val = (s[t][g] + bw[i_l * 48 + rel] + s_rbias[mc]) * scale;
        if ((J + j_l) > (Iw + i_l) + cMLEN) val = -3.4028e38f;
        s[t][g] = val;
      }
    }

    // -------- online softmax: row reductions over the 16-lane half-groups
    float alpha[8];
#pragma unroll
    for (int g = 0; g < 8; ++g) {
      float v = fmaxf(s[0][g], s[1][g]);
      v = fmaxf(v, __shfl_xor(v, 1, 16));
      v = fmaxf(v, __shfl_xor(v, 2, 16));
      v = fmaxf(v, __shfl_xor(v, 4, 16));
      v = fmaxf(v, __shfl_xor(v, 8, 16));
      const float mnew = fmaxf(mrow[g], v);
      alpha[g] = __expf(mrow[g] - mnew);
      mrow[g] = mnew;
    }
#pragma unroll
    for (int g = 0; g < 8; ++g) {
      const float p0 = __expf(s[0][g] - mrow[g]);
      const float p1 = __expf(s[1][g] - mrow[g]);
      s[0][g] = p0; s[1][g] = p1;
      float r = p0 + p1;
      r += __shfl_xor(r, 1, 16);
      r += __shfl_xor(r, 2, 16);
      r += __shfl_xor(r, 4, 16);
      r += __shfl_xor(r, 8, 16);
      lrow[g] = lrow[g] * alpha[g] + r;
    }

    // -------- redistribute P through LDS into A-operand layout
#pragma unroll
    for (int t = 0; t < 2; ++t)
#pragma unroll
      for (int g = 0; g < 8; ++g)
        pbuf[(g + 8 * h) * 32 + 16 * t + cl] = (__bf16)s[t][g];
    asm volatile("s_wait_dscnt 0x0" ::: "memory");

    const v16bf pa = wmma_op_load_bf16(pbuf, 32);  // A: 16 x 32 (K = j)
#pragma unroll
    for (int u = 0; u < 4; ++u) {
#pragma unroll
      for (int g = 0; g < 8; ++g) O[u][g] *= alpha[g];
      const bf16* vt = vtb + (size_t)(16 * u) * cKLEN + J;  // [N=d][K=j]
      O[u] = wmma_bf16(pa, wmma_op_load_bf16(vt, cKLEN), O[u]);
    }
  }

  // -------- finalize: O /= rowsum, write attn_vec bf16 at [(i*B+b)][n*64+d]
#pragma unroll
  for (int u = 0; u < 4; ++u)
#pragma unroll
    for (int g = 0; g < 8; ++g) {
      const int i_l = g + 8 * h;
      const int d = 16 * u + cl;
      const float val = O[u][g] / lrow[g];
      avec[((size_t)(Iw + i_l) * cBSZ + b) * cDM + n * cDH + d] = (__bf16)val;
    }
}

// ---------------------------------------------------------------------------
// Kernel 5: output projection.  A = avec bf16 [4096][1024], B = WoT bf16
// [1024][1024] -> attn_out fp32 [4096][1024].
// ---------------------------------------------------------------------------
__global__ __launch_bounds__(128) void out_gemm_kernel(
    const bf16* __restrict__ avec, const bf16* __restrict__ woT,
    float* __restrict__ attn_out) {
  const int wid = blockIdx.x * 4 + (threadIdx.x >> 5);
  const int nt = wid % (cDM / 64);
  const int mt = wid / (cDM / 64);
  const int r0 = mt * 16, c0 = nt * 64;
  const int lane = threadIdx.x & 31, cl = lane & 15, h = lane >> 4;

  v8f acc[4];
#pragma unroll
  for (int u = 0; u < 4; ++u) acc[u] = vzero8();

  for (int kk = 0; kk < cDM; kk += 32) {
    v16bf a = wmma_op_load_bf16(avec + (size_t)r0 * cDM + kk, cDM);
#pragma unroll
    for (int u = 0; u < 4; ++u) {
      v16bf bb = wmma_op_load_bf16(woT + (size_t)(c0 + 16 * u) * cDM + kk, cDM);
      acc[u] = wmma_bf16(a, bb, acc[u]);
    }
  }
#pragma unroll
  for (int u = 0; u < 4; ++u)
#pragma unroll
    for (int g = 0; g < 8; ++g)
      attn_out[(size_t)(r0 + g + 8 * h) * cDM + c0 + 16 * u + cl] = acc[u][g];
}

// ---------------------------------------------------------------------------
// Kernel 6: residual + layernorm.  One 256-thread block per (i,b) row.
// ---------------------------------------------------------------------------
__global__ __launch_bounds__(256) void ln_kernel(
    const float* __restrict__ w, const float* __restrict__ attn_out,
    const float* __restrict__ g, const float* __restrict__ bb,
    float* __restrict__ out) {
  __shared__ float red[256];
  const int row = blockIdx.x;
  const int base = threadIdx.x * 4;
  const float* wr = w + (size_t)row * cDM;
  const float* ar = attn_out + (size_t)row * cDM;

  float x[4]; float s = 0.0f;
#pragma unroll
  for (int e = 0; e < 4; ++e) { x[e] = wr[base + e] + ar[base + e]; s += x[e]; }
  red[threadIdx.x] = s; __syncthreads();
  for (int o = 128; o > 0; o >>= 1) {
    if (threadIdx.x < o) red[threadIdx.x] += red[threadIdx.x + o];
    __syncthreads();
  }
  const float mu = red[0] * (1.0f / cDM);
  __syncthreads();
  float vs = 0.0f;
#pragma unroll
  for (int e = 0; e < 4; ++e) { const float d = x[e] - mu; vs += d * d; }
  red[threadIdx.x] = vs; __syncthreads();
  for (int o = 128; o > 0; o >>= 1) {
    if (threadIdx.x < o) red[threadIdx.x] += red[threadIdx.x + o];
    __syncthreads();
  }
  const float rinv = rsqrtf(red[0] * (1.0f / cDM) + 1e-5f);
#pragma unroll
  for (int e = 0; e < 4; ++e)
    out[(size_t)row * cDM + base + e] =
        (x[e] - mu) * rinv * g[base + e] + bb[base + e];
}

// ---------------------------------------------------------------------------
// Launch
// ---------------------------------------------------------------------------
extern "C" void kernel_launch(void* const* d_in, const int* in_sizes, int n_in,
                              void* d_out, int out_size, void* d_ws, size_t ws_size,
                              hipStream_t stream) {
  const float* w        = (const float*)d_in[0];
  const float* mems     = (const float*)d_in[1];
  const float* r_emb    = (const float*)d_in[2];
  const float* r_w_bias = (const float*)d_in[3];
  const float* r_bias   = (const float*)d_in[4];
  const float* Wqkv     = (const float*)d_in[5];
  const float* Wo       = (const float*)d_in[6];
  const float* ln_g     = (const float*)d_in[7];
  const float* ln_b     = (const float*)d_in[8];
  // d_in[9] = attn_mask (bool): causal structure handled analytically.
  float* out = (float*)d_out;

  char* ws = (char*)d_ws;
  size_t off = 0;
  auto alloc = [&](size_t bytes) -> void* {
    void* p = ws + off;
    off += (bytes + 255) & ~(size_t)255;
    return p;
  };
  bf16* wqkvT   = (bf16*)alloc((size_t)cQKV * cDM * 2);               // 6 MB
  bf16* woT     = (bf16*)alloc((size_t)cDM * cDM * 2);                // 2 MB
  bf16* remb_bf = (bf16*)alloc((size_t)cKLEN * cNH * cDH * 2);        // 4 MB
  bf16* q_bf    = (bf16*)alloc((size_t)cBSZ * cNH * cQLEN * cDH * 2); // 8 MB
  bf16* rwq_bf  = (bf16*)alloc((size_t)cBSZ * cNH * cQLEN * cDH * 2); // 8 MB
  bf16* k_bf    = (bf16*)alloc((size_t)cBSZ * cNH * cKLEN * cDH * 2); // 16 MB
  bf16* vT_bf   = (bf16*)alloc((size_t)cBSZ * cNH * cDH * cKLEN * 2); // 16 MB
  bf16* avec    = (bf16*)alloc((size_t)cQLEN * cBSZ * cDM * 2);       // 8 MB
  float* attn_o = (float*)alloc((size_t)cQLEN * cBSZ * cDM * 4);      // 16 MB
  (void)ws_size; (void)in_sizes; (void)n_in; (void)out_size;

  // 1) weight transpose+convert (independent)
  transpose_cvt_kernel<<<dim3(cQKV / 32, cDM / 32), 256, 0, stream>>>(Wqkv, wqkvT, cDM, cQKV);
  transpose_cvt_kernel<<<dim3(cDM / 32, cDM / 32), 256, 0, stream>>>(Wo, woT, cDM, cDM);
  cvt_bf16_kernel<<<(cKLEN * cNH * cDH) / 256, 256, 0, stream>>>(r_emb, remb_bf, cKLEN * cNH * cDH);

  // 2) QKV GEMM: 512 M-tiles x 48 N-tiles = 24576 waves -> 6144 blocks
  qkv_gemm_kernel<<<6144, 128, 0, stream>>>(w, mems, wqkvT, r_w_bias,
                                            q_bf, rwq_bf, k_bf, vT_bf);

  // 3) flash attention: BSZ*NH*(QLEN/64) = 1024 blocks, 4 waves each
  attn_kernel<<<1024, 128, 0, stream>>>(q_bf, rwq_bf, k_bf, vT_bf,
                                        remb_bf, r_bias, avec);

  // 4) output projection: 256 x 16 = 4096 waves -> 1024 blocks
  out_gemm_kernel<<<1024, 128, 0, stream>>>(avec, woT, attn_o);

  // 5) residual + layernorm: one block per (i,b) row
  ln_kernel<<<cQLEN * cBSZ, 256, 0, stream>>>(w, attn_o, ln_g, ln_b, out);
}
